// MultiHeadAttention_49984829391411
// MI455X (gfx1250) — matrix operand order repaired and checked by reading, not compiled, not run on hardware
//
#include <hip/hip_runtime.h>

typedef __attribute__((ext_vector_type(16))) __bf16 v16bf;
typedef __attribute__((ext_vector_type(8)))  float  v8f;
typedef __attribute__((ext_vector_type(4)))  unsigned uint4v;
typedef __attribute__((ext_vector_type(8)))  int      int8v;
typedef __attribute__((ext_vector_type(4)))  int      int4v;

union Frag { unsigned u[8]; v16bf v; };

static constexpr int Bc = 2, Sc = 2048, Dc = 1024, Hc = 16, DKc = 64;

#if __has_builtin(__builtin_amdgcn_tensor_load_to_lds) && \
    __has_builtin(__builtin_amdgcn_s_wait_tensorcnt)
#define USE_TDM 1
#else
#define USE_TDM 0
#endif

__device__ __forceinline__ unsigned short f2bf(float f) {
  unsigned u = __builtin_bit_cast(unsigned, f);
  return (unsigned short)((u + 0x7FFFu + ((u >> 16) & 1u)) >> 16);
}
__device__ __forceinline__ unsigned pack2(float a, float b) {
  return (unsigned)f2bf(a) | ((unsigned)f2bf(b) << 16);
}
__device__ __forceinline__ v8f wmma_bf16(const Frag& a, const Frag& b, v8f c) {
  return __builtin_amdgcn_wmma_f32_16x16x32_bf16(false, a.v, false, b.v,
                                                 (short)0, c, false, false);
}
__device__ __forceinline__ unsigned lds_off(const void* p) {
  return (unsigned)(size_t)p;  // low 32 bits of generic ptr == LDS byte offset
}

// ---------------------------------------------------------------------------
// GEMM: Out[m,n] = sum_k A[m,k] * W[n,k] + bias[n]   (y = x @ W^T + b)
// M=4096, N=1024, K=1024. Block tile 128x128x32, 256 thr = 8 waves,
// each wave computes 32x64 (2x4 WMMA accumulators).
// A_F32=false path stages the A tile with global_load_async_to_lds_b128.
// ---------------------------------------------------------------------------
template <bool A_F32, bool OUT_HEADS>
__global__ __launch_bounds__(256) void wmma_gemm(const void* __restrict__ Ap,
                                                 const float* __restrict__ W,
                                                 const float* __restrict__ bias,
                                                 void* __restrict__ Out) {
  constexpr int LP = 40;  // LDS pitch (bf16): 80 B rows -> 16B-aligned chunks
  __shared__ unsigned short Al[128 * LP];
  __shared__ unsigned short Bl[128 * LP];

  const int t = threadIdx.x;
  const int m0 = blockIdx.x * 128;
  const int n0 = blockIdx.y * 128;
  const int wave = t >> 5, lane = t & 31;
  const int half = lane >> 4, l16 = lane & 15;
  const int wm = wave & 3, wn = wave >> 2;

  v8f acc[2][4];
#pragma unroll
  for (int i = 0; i < 2; ++i)
#pragma unroll
    for (int j = 0; j < 4; ++j) acc[i][j] = {};

  for (int k0 = 0; k0 < Dc; k0 += 32) {
    // ---- stage A tile (128x32 bf16) ----
    if constexpr (A_F32) {
#pragma unroll
      for (int i = 0; i < 4; ++i) {
        const int slot = t + i * 256;
        const int row = slot >> 3, kc = (slot & 7) * 4;
        const float4 x =
            *(const float4*)((const float*)Ap + (size_t)(m0 + row) * Dc + k0 + kc);
        unsigned short* p = &Al[row * LP + kc];
        *(unsigned*)(p) = pack2(x.x, x.y);
        *(unsigned*)(p + 2) = pack2(x.z, x.w);
      }
    } else {
      // async DMA: each slot copies 16 B global -> LDS (ASYNCcnt path)
#pragma unroll
      for (int i = 0; i < 2; ++i) {
        const int slot = t + i * 256;  // 512 slots x 8 bf16
        const int row = slot >> 2, kc = (slot & 3) * 8;
        const unsigned dst = lds_off(&Al[row * LP + kc]);
        const unsigned short* g =
            (const unsigned short*)Ap + (size_t)(m0 + row) * Dc + k0 + kc;
        asm volatile("global_load_async_to_lds_b128 %0, %1, off"
                     :: "v"(dst), "v"(g) : "memory");
      }
    }
    // ---- stage B tile: rows are output columns n (W is [N,K] row-major) ----
#pragma unroll
    for (int i = 0; i < 4; ++i) {
      const int slot = t + i * 256;
      const int row = slot >> 3, kc = (slot & 7) * 4;
      const float4 x = *(const float4*)(W + (size_t)(n0 + row) * Dc + k0 + kc);
      unsigned short* p = &Bl[row * LP + kc];
      *(unsigned*)(p) = pack2(x.x, x.y);
      *(unsigned*)(p + 2) = pack2(x.z, x.w);
    }
    if (k0 + 32 < Dc) {  // prefetch next k-tile -> global_prefetch_b8
      __builtin_prefetch((const char*)Ap + ((size_t)(m0 + (t >> 1)) * Dc + k0 + 32) *
                                               (A_F32 ? 4 : 2), 0, 0);
      __builtin_prefetch(W + (size_t)(n0 + (t >> 1)) * Dc + k0 + 32, 0, 0);
    }
    if constexpr (!A_F32) asm volatile("s_wait_asynccnt 0" ::: "memory");
    __syncthreads();

    // ---- load fragments, contiguous k = 16*half + s for A and B ----
    Frag a[2];
#pragma unroll
    for (int i = 0; i < 2; ++i) {
      const int row = wm * 32 + i * 16 + l16;
#pragma unroll
      for (int v = 0; v < 8; ++v) {
        const int k = 2 * v + half * 16;
        a[i].u[v] = *(const unsigned*)&Al[row * LP + k];
      }
    }
    Frag b[4];
#pragma unroll
    for (int j = 0; j < 4; ++j) {
      const int col = wn * 64 + j * 16 + l16;
#pragma unroll
      for (int v = 0; v < 8; ++v) {
        const int k = 2 * v + half * 16;
        b[j].u[v] = *(const unsigned*)&Bl[col * LP + k];
      }
    }
#pragma unroll
    for (int i = 0; i < 2; ++i)
#pragma unroll
      for (int j = 0; j < 4; ++j) acc[i][j] = wmma_bf16(a[i], b[j], acc[i][j]);
    __syncthreads();
  }

  // ---- epilogue: bias + store (C layout: reg r -> M=r+8*half, N=l16) ----
#pragma unroll
  for (int i = 0; i < 2; ++i) {
#pragma unroll
    for (int j = 0; j < 4; ++j) {
      const int n = n0 + wn * 64 + j * 16 + l16;
      const float bn = bias[n];
#pragma unroll
      for (int r = 0; r < 8; ++r) {
        const int m = m0 + wm * 32 + i * 16 + r + half * 8;
        const float val = acc[i][j][r] + bn;
        if (OUT_HEADS) {
          const int bb = m >> 11, ss = m & (Sc - 1);
          const int hh = n >> 6, dk = n & 63;
          ((unsigned short*)Out)[(((size_t)(bb * Hc + hh) * Sc + ss) << 6) + dk] =
              f2bf(val);
        } else {
          ((float*)Out)[(size_t)m * Dc + n] = val;
        }
      }
    }
  }
}

// ---------------------------------------------------------------------------
// Flash attention. Grid: (B*H, S/128). 8 waves; each wave owns 16 query rows.
// K tile staged by the Tensor Data Mover (with D# LDS padding recreating the
// 66-element pitch); V staged manually (transposed). TENSORcnt drained by the
// issuing wave before the workgroup barrier.
// ---------------------------------------------------------------------------
__global__ __launch_bounds__(256) void flash_attn(
    const unsigned short* __restrict__ Q, const unsigned short* __restrict__ K,
    const unsigned short* __restrict__ V, const int* __restrict__ mask,
    unsigned short* __restrict__ O) {
  constexpr int KP = 66;  // pitch (even) for 64-wide tiles
  __shared__ unsigned short Kl[64 * KP];      // [key][dk]
  __shared__ unsigned short Vt[64 * KP];      // [dk][key]  (transposed)
  __shared__ unsigned short Pl[8][16 * 34];   // per-wave P bounce tile 16x32

  const int t = threadIdx.x;
  const int wave = t >> 5, lane = t & 31, half = lane >> 4, l16 = lane & 15;
  const int bh = blockIdx.x;
  const int b = bh >> 4, h = bh & 15;
  const size_t base = (size_t)bh * Sc * DKc;
  const int qr0 = blockIdx.y * 128 + wave * 16;

  // preload Q fragments (A-matrix 16x32 x2 k-steps) straight from global
  Frag qf[2];
#pragma unroll
  for (int ks = 0; ks < 2; ++ks)
#pragma unroll
    for (int v = 0; v < 8; ++v) {
      const int k = 2 * v + half * 16 + ks * 32;
      qf[ks].u[v] = *(const unsigned*)(Q + base + (size_t)(qr0 + l16) * DKc + k);
    }

  float mrow[8], lsum[8];
  v8f o[4];
#pragma unroll
  for (int r = 0; r < 8; ++r) { mrow[r] = -3.0e38f; lsum[r] = 0.0f; }
#pragma unroll
  for (int nt = 0; nt < 4; ++nt) o[nt] = {};

  for (int kb = 0; kb < Sc / 64; ++kb) {
    const int key0 = kb * 64;
#if USE_TDM
    if (wave == 0) {
      // ---- D# for a 64(dk) x 64(key) bf16 tile, LDS pitch 66 via padding ----
      const unsigned long long ga =
          (unsigned long long)(size_t)(K + base + (size_t)key0 * DKc);
      uint4v g0;
      g0.x = 1u;                    // count=1 (valid user descriptor)
      g0.y = lds_off(Kl);           // lds_addr
      g0.z = (unsigned)ga;          // global_addr[31:0]
      g0.w = (unsigned)((ga >> 32) & 0x01FFFFFFu) | 0x80000000u;  // addr|type=2
      int8v g1 = {0, 0, 0, 0, 0, 0, 0, 0};
      g1[0] = (int)((1u << 16)      // data_size = 2 bytes
                    | (1u << 20)    // pad_enable
                    | (4u << 22));  // pad_interval: every 32 dwords (one row)
                                    // pad_amount code 0 = 1 dword -> pitch 66
      g1[1] = (int)(64u << 16);     // tensor_dim0 = 64 (dk)
      g1[2] = (int)(2048u << 16);   // tensor_dim1 = 2048 (keys in head)
      g1[3] = (int)(64u << 16);     // tile_dim0 = 64
      g1[4] = 64;                   // tile_dim1 = 64 keys
      g1[5] = 64;                   // tensor_dim0_stride = 64 elements
      const int4v gz4 = {0, 0, 0, 0};
      const int8v gz8 = {0, 0, 0, 0, 0, 0, 0, 0};
      __builtin_amdgcn_tensor_load_to_lds(g0, g1, gz4, gz4, gz8, 0);
    }
#else
#pragma unroll
    for (int i = 0; i < 4; ++i) {
      const int slot = t + i * 256;
      const int key = slot >> 4, dkc = (slot & 15) * 4;
      const uint2 x = *(const uint2*)(K + base + (size_t)(key0 + key) * DKc + dkc);
      unsigned short* p = &Kl[key * KP + dkc];
      *(unsigned*)p = x.x;
      *(unsigned*)(p + 2) = x.y;
    }
#endif
    // ---- stage V transposed (manual; TDM has no transpose) ----
#pragma unroll
    for (int i = 0; i < 4; ++i) {
      const int slot = t + i * 256;
      const int key = slot >> 4, dkc = (slot & 15) * 4;
      const uint2 y = *(const uint2*)(V + base + (size_t)(key0 + key) * DKc + dkc);
      Vt[(dkc + 0) * KP + key] = (unsigned short)(y.x & 0xFFFF);
      Vt[(dkc + 1) * KP + key] = (unsigned short)(y.x >> 16);
      Vt[(dkc + 2) * KP + key] = (unsigned short)(y.y & 0xFFFF);
      Vt[(dkc + 3) * KP + key] = (unsigned short)(y.y >> 16);
    }
#if USE_TDM
    if (wave == 0) __builtin_amdgcn_s_wait_tensorcnt(0);
#endif
    __syncthreads();

#pragma unroll
    for (int sb = 0; sb < 2; ++sb) {
      // ---- scores: S = Q*K^T (two 16x16 tiles covering 32 keys) ----
      v8f s[2];
#pragma unroll
      for (int j = 0; j < 2; ++j) {
        v8f sj = {};
        const int keyc = sb * 32 + j * 16 + l16;
#pragma unroll
        for (int ks = 0; ks < 2; ++ks) {
          Frag bb;
#pragma unroll
          for (int v = 0; v < 8; ++v) {
            const int dk = 2 * v + half * 16 + ks * 32;
            bb.u[v] = *(const unsigned*)&Kl[keyc * KP + dk];
          }
          sj = wmma_bf16(qf[ks], bb, sj);
        }
        const int keyg = key0 + sb * 32 + j * 16 + l16;
        const float addm = (mask[(b << 11) + keyg] == 0) ? -1e9f : 0.0f;
#pragma unroll
        for (int r = 0; r < 8; ++r) sj[r] = sj[r] * 0.125f + addm;  // 1/sqrt(64)
        s[j] = sj;
      }
      // ---- online softmax (row = r + 8*half, cols across 16-lane group) ----
      float alpha[8];
#pragma unroll
      for (int r = 0; r < 8; ++r) {
        float mx = fmaxf(s[0][r], s[1][r]);
        mx = fmaxf(mx, __shfl_xor(mx, 8));
        mx = fmaxf(mx, __shfl_xor(mx, 4));
        mx = fmaxf(mx, __shfl_xor(mx, 2));
        mx = fmaxf(mx, __shfl_xor(mx, 1));
        const float mn = fmaxf(mrow[r], mx);
        alpha[r] = __expf(mrow[r] - mn);
        mrow[r] = mn;
        const float p0 = __expf(s[0][r] - mn);
        const float p1 = __expf(s[1][r] - mn);
        s[0][r] = p0;
        s[1][r] = p1;
        float ps = p0 + p1;
        ps += __shfl_xor(ps, 8);
        ps += __shfl_xor(ps, 4);
        ps += __shfl_xor(ps, 2);
        ps += __shfl_xor(ps, 1);
        lsum[r] = lsum[r] * alpha[r] + ps;
#pragma unroll
        for (int nt = 0; nt < 4; ++nt) o[nt][r] *= alpha[r];
      }
      // ---- bounce P through per-wave LDS to reach A-fragment layout ----
#pragma unroll
      for (int j = 0; j < 2; ++j)
#pragma unroll
        for (int r = 0; r < 8; ++r)
          Pl[wave][(r + 8 * half) * 34 + j * 16 + l16] = f2bf(s[j][r]);
      Frag pa;
#pragma unroll
      for (int v = 0; v < 8; ++v) {
        const int k = 2 * v + half * 16;
        pa.u[v] = *(const unsigned*)&Pl[wave][l16 * 34 + k];
      }
      // ---- O += P * V  (A=P 16x32, B=V 32x16 per dk tile) ----
#pragma unroll
      for (int nt = 0; nt < 4; ++nt) {
        Frag bv;
        const int dkr = nt * 16 + l16;
#pragma unroll
        for (int v = 0; v < 8; ++v) {
          const int keyl = sb * 32 + 2 * v + half * 16;
          bv.u[v] = *(const unsigned*)&Vt[dkr * KP + keyl];
        }
        o[nt] = wmma_bf16(pa, bv, o[nt]);
      }
    }
    __syncthreads();
  }

  // ---- finalize: O/l, store bf16 merged-head [B,S,D] ----
  float invl[8];
#pragma unroll
  for (int r = 0; r < 8; ++r) invl[r] = 1.0f / lsum[r];
#pragma unroll
  for (int nt = 0; nt < 4; ++nt)
#pragma unroll
    for (int r = 0; r < 8; ++r) {
      const int ss = qr0 + r + 8 * half;
      const int dk = h * 64 + nt * 16 + l16;
      O[((size_t)(b * Sc + ss)) * Dc + dk] = f2bf(o[nt][r] * invl[r]);
    }
}

// ---------------------------------------------------------------------------
extern "C" void kernel_launch(void* const* d_in, const int* in_sizes, int n_in,
                              void* d_out, int out_size, void* d_ws, size_t ws_size,
                              hipStream_t stream) {
  const float* q = (const float*)d_in[0];
  const float* k = (const float*)d_in[1];
  const float* v = (const float*)d_in[2];
  const int* mk = (const int*)d_in[3];
  const float* wq = (const float*)d_in[4];
  const float* bq = (const float*)d_in[5];
  const float* wk = (const float*)d_in[6];
  const float* bk = (const float*)d_in[7];
  const float* wv = (const float*)d_in[8];
  const float* bv = (const float*)d_in[9];
  const float* wo = (const float*)d_in[10];
  const float* bo = (const float*)d_in[11];

  const size_t perT = (size_t)Bc * Hc * Sc * DKc;  // 4Mi bf16 elems = 8MB
  unsigned short* Qw = (unsigned short*)d_ws;
  unsigned short* Kw = Qw + perT;
  unsigned short* Vw = Kw + perT;
  unsigned short* Ow = Vw + perT;

  const dim3 gridG((Bc * Sc) / 128, Dc / 128);  // 32 x 8
  wmma_gemm<true, true><<<gridG, 256, 0, stream>>>(q, wq, bq, Qw);
  wmma_gemm<true, true><<<gridG, 256, 0, stream>>>(k, wk, bk, Kw);
  wmma_gemm<true, true><<<gridG, 256, 0, stream>>>(v, wv, bv, Vw);
  flash_attn<<<dim3(Bc * Hc, Sc / 128), 256, 0, stream>>>(Qw, Kw, Vw, mk, Ow);
  wmma_gemm<false, false><<<gridG, 256, 0, stream>>>(Ow, wo, bo, d_out);
}
